// VoxelResBlock_49478023250388
// MI455X (gfx1250) — matrix operand order repaired and checked by reading, not compiled, not run on hardware
//
#include <hip/hip_runtime.h>

#define BN_EPS 1e-5f

typedef __attribute__((ext_vector_type(16))) _Float16 v16h;
typedef __attribute__((ext_vector_type(8)))  _Float16 v8h;
typedef __attribute__((ext_vector_type(4)))  _Float16 v4h;
typedef __attribute__((ext_vector_type(8)))  float    v8f;

// Load a 16-bit A-fragment slice for one lane: elements 0..15 = p[0..15]
// (K = base..base+15).
static __device__ __forceinline__ v16h load_a16(const _Float16* p) {
  v8h lo = *(const v8h*)(p);
  v8h hi = *(const v8h*)(p + 8);
  return __builtin_shufflevector(lo, hi, 0,1,2,3,4,5,6,7,8,9,10,11,12,13,14,15);
}

// ---------------------------------------------------------------------------
// Pre-swizzle weights W[K][chunks*32][64] (f32) into per-lane WMMA B-operand
// fragments (f16). Fragment f = (k*chunks + ch)*4 + nt is a 32x16 (KxN) tile:
// lane L holds column nt*16 + (L&15), K-values ch*32 + ((L&16)?16:0) + 0..15,
// stored as 16 contiguous halves at frags[f*512 + L*16].
// ---------------------------------------------------------------------------
__global__ void prep_frags(const float* __restrict__ W, _Float16* __restrict__ frags,
                           int K, int chunks) {
  int t = blockIdx.x * blockDim.x + threadIdx.x;
  int total = K * chunks * 4 * 32;
  if (t >= total) return;
  int lane = t & 31;
  int frag = t >> 5;
  int nt = frag & 3;
  int ch = (frag >> 2) % chunks;
  int k  = frag / (4 * chunks);
  int col = nt * 16 + (lane & 15);
  int kb  = ch * 32 + ((lane & 16) ? 16 : 0);
  int cin = chunks * 32;
  const float* wsrc = W + ((size_t)k * cin + kb) * 64 + col;
  _Float16* dst = frags + (size_t)frag * 512 + (size_t)lane * 16;
#pragma unroll
  for (int j = 0; j < 16; ++j)
    dst[j] = (_Float16)wsrc[(size_t)j * 64];
}

// f32 -> f16 cast, 4 elements per thread
__global__ void f32_to_f16(const float* __restrict__ src, _Float16* __restrict__ dst,
                           int total4) {
  int t = blockIdx.x * blockDim.x + threadIdx.x;
  if (t >= total4) return;
  float4 v = ((const float4*)src)[t];
  v4h o;
  o[0] = (_Float16)v.x; o[1] = (_Float16)v.y;
  o[2] = (_Float16)v.z; o[3] = (_Float16)v.w;
  ((v4h*)dst)[t] = o;
}

// ---------------------------------------------------------------------------
// WMMA sparse-conv GEMM: out[i,:64] = sum_k feats[nbr[k][i], :] @ Wk
// One wave handles 32 rows (2 row-tiles) x 64 cols (4 N-tiles).
// Weight fragments for the current k are staged in LDS (double-buffered,
// loaded cooperatively by the whole block once per k -> 8x less L2 traffic).
// CHUNKS = input channels / 32. nbr == nullptr => identity row mapping.
// ---------------------------------------------------------------------------
template <int CHUNKS>
__global__ __launch_bounds__(256) void conv_wmma(
    const _Float16* __restrict__ feats, int fstride,
    const int* __restrict__ nbr, int K,
    const _Float16* __restrict__ frags,
    float* __restrict__ out, int N) {
  // One k's fragments: CHUNKS*4 frags * 1024 B = CHUNKS*4096 B = CHUNKS*256 uint4
  __shared__ uint4 smem[2][CHUNKS * 256];

  const int tid  = threadIdx.x;
  const int lane = tid & 31;
  const int gwave = blockIdx.x * 8 + (tid >> 5);
  const int base  = gwave * 32;
  const bool active = base < N;      // wave-uniform; inactive waves still barrier
  const int l15 = lane & 15;
  const int hi  = (lane >> 4) & 1;

  const uint4* gfrags = (const uint4*)frags;   // per-k stride = CHUNKS*256 uint4

  // Preload k=0 fragments into buffer 0 (visible after first barrier).
#pragma unroll
  for (int c = 0; c < CHUNKS; ++c)
    smem[0][c * 256 + tid] = gfrags[c * 256 + tid];

  v8f acc0[4] = {};   // rows [base, base+16)
  v8f acc1[4] = {};   // rows [base+16, base+32)

  for (int k = 0; k < K; ++k) {
    __syncthreads();                 // k's buffer is now fully written
    const int b = k & 1;

    // Kick off next k's fragment fetch into registers (overlaps the WMMAs).
    uint4 stage[CHUNKS];
    if (k + 1 < K) {
#pragma unroll
      for (int c = 0; c < CHUNKS; ++c)
        stage[c] = gfrags[(size_t)(k + 1) * CHUNKS * 256 + c * 256 + tid];
      if (k + 2 < K && nbr)          // global_prefetch for the next index slab
        __builtin_prefetch(nbr + (size_t)(k + 2) * N + base + l15, 0, 0);
    }

    // Row gather indices for this offset.
    int r0 = 0, r1 = 0;
    if (active) {
      if (nbr) {
        const int* nb = nbr + (size_t)k * N + base;
        r0 = nb[l15];
        r1 = nb[16 + l15];
      } else {
        r0 = base + l15;       if (r0 >= N) r0 = N - 1;
        r1 = base + 16 + l15;  if (r1 >= N) r1 = N - 1;
      }
    }

    const uint4* sb = smem[b];
#pragma unroll
    for (int ch = 0; ch < CHUNKS; ++ch) {
      v16h a0 = load_a16(feats + (size_t)r0 * fstride + ch * 32 + hi * 16);
      v16h a1 = load_a16(feats + (size_t)r1 * fstride + ch * 32 + hi * 16);
#pragma unroll
      for (int nt = 0; nt < 4; ++nt) {
        // Lane's 32 contiguous bytes of B fragment (ch,nt) from LDS.
        v16h bf = *(const v16h*)(sb + (ch * 4 + nt) * 64 + lane * 2);
        acc0[nt] = __builtin_amdgcn_wmma_f32_16x16x32_f16(
            false, a0, false, bf, (short)0, acc0[nt], false, false);
        acc1[nt] = __builtin_amdgcn_wmma_f32_16x16x32_f16(
            false, a1, false, bf, (short)0, acc1[nt], false, false);
      }
    }

    // Publish next k's fragments into the other buffer.
    if (k + 1 < K) {
#pragma unroll
      for (int c = 0; c < CHUNKS; ++c)
        smem[b ^ 1][c * 256 + tid] = stage[c];
    }
  }

  if (!active) return;
  // C/D layout: lane holds col (nt*16 + l15); VGPR j is row hi*8 + j.
  const int row0 = base + hi * 8;
#pragma unroll
  for (int nt = 0; nt < 4; ++nt) {
    int col = nt * 16 + l15;
#pragma unroll
    for (int j = 0; j < 8; ++j) {
      int ra = row0 + j;
      int rb = row0 + 16 + j;
      if (ra < N) out[(size_t)ra * 64 + col] = acc0[nt][j];
      if (rb < N) out[(size_t)rb * 64 + col] = acc1[nt][j];
    }
  }
}

// ---------------------------------------------------------------------------
// Deterministic BN statistics: per-block partial sums / sums-of-squares.
// ---------------------------------------------------------------------------
__global__ void bn_stats_partial(const float* __restrict__ src,
                                 float* __restrict__ partials, int N) {
  __shared__ float ssum[256];
  __shared__ float ssq[256];
  int tid = threadIdx.x;
  int c = tid & 63;
  int rsub = tid >> 6;
  float sum = 0.f, sq = 0.f;
  for (int row = blockIdx.x * 4 + rsub; row < N; row += gridDim.x * 4) {
    float v = src[(size_t)row * 64 + c];
    sum += v;
    sq  += v * v;
  }
  ssum[tid] = sum; ssq[tid] = sq;
  __syncthreads();
  if (tid < 64) {
    float s = ssum[tid] + ssum[tid + 64] + ssum[tid + 128] + ssum[tid + 192];
    float q = ssq[tid]  + ssq[tid + 64]  + ssq[tid + 128]  + ssq[tid + 192];
    partials[(size_t)blockIdx.x * 128 + tid] = s;
    partials[(size_t)blockIdx.x * 128 + 64 + tid] = q;
  }
}

// Reduce partials, produce fused scale/shift: ab[c]=g*rsqrt(v+eps), ab[64+c]=b-m*a
__global__ void bn_finalize(const float* __restrict__ partials, int nparts,
                            const float* __restrict__ g, const float* __restrict__ b,
                            float* __restrict__ ab, float invN) {
  int c = threadIdx.x;
  if (c >= 64) return;
  float s = 0.f, q = 0.f;
  for (int p = 0; p < nparts; ++p) {
    s += partials[(size_t)p * 128 + c];
    q += partials[(size_t)p * 128 + 64 + c];
  }
  float m = s * invN;
  float v = q * invN - m * m;
  float a = g[c] * rsqrtf(v + BN_EPS);
  ab[c] = a;
  ab[64 + c] = b[c] - m * a;
}

// h = f16(relu(bn(y))), 4 elems/thread (row-major, 64 channels)
__global__ void bn_relu_f16(const float* __restrict__ y, const float* __restrict__ ab,
                            _Float16* __restrict__ h, int total4) {
  int t = blockIdx.x * blockDim.x + threadIdx.x;
  if (t >= total4) return;
  int c0 = (t * 4) & 63;
  float4 v = ((const float4*)y)[t];
  v4h o;
  float r;
  r = v.x * ab[c0 + 0] + ab[64 + c0 + 0]; o[0] = (_Float16)fmaxf(r, 0.f);
  r = v.y * ab[c0 + 1] + ab[64 + c0 + 1]; o[1] = (_Float16)fmaxf(r, 0.f);
  r = v.z * ab[c0 + 2] + ab[64 + c0 + 2]; o[2] = (_Float16)fmaxf(r, 0.f);
  r = v.w * ab[c0 + 3] + ab[64 + c0 + 3]; o[3] = (_Float16)fmaxf(r, 0.f);
  ((v4h*)h)[t] = o;
}

// out = relu(bn2(y2) + bn_id(ident)), 4 elems/thread
__global__ void final_out_k(const float* __restrict__ y2, const float* __restrict__ id,
                            const float* __restrict__ ab2, const float* __restrict__ abid,
                            float* __restrict__ out, int total4) {
  int t = blockIdx.x * blockDim.x + threadIdx.x;
  if (t >= total4) return;
  int c0 = (t * 4) & 63;
  float4 a = ((const float4*)y2)[t];
  float4 b = ((const float4*)id)[t];
  float4 o;
  o.x = fmaxf(a.x * ab2[c0 + 0] + ab2[64 + c0 + 0] + b.x * abid[c0 + 0] + abid[64 + c0 + 0], 0.f);
  o.y = fmaxf(a.y * ab2[c0 + 1] + ab2[64 + c0 + 1] + b.y * abid[c0 + 1] + abid[64 + c0 + 1], 0.f);
  o.z = fmaxf(a.z * ab2[c0 + 2] + ab2[64 + c0 + 2] + b.z * abid[c0 + 2] + abid[64 + c0 + 2], 0.f);
  o.w = fmaxf(a.w * ab2[c0 + 3] + ab2[64 + c0 + 3] + b.w * abid[c0 + 3] + abid[64 + c0 + 3], 0.f);
  ((float4*)out)[t] = o;
}

extern "C" void kernel_launch(void* const* d_in, const int* in_sizes, int n_in,
                              void* d_out, int out_size, void* d_ws, size_t ws_size,
                              hipStream_t stream) {
  const float* x   = (const float*)d_in[0];
  const int*   nbr = (const int*)d_in[1];
  const float* w1  = (const float*)d_in[2];
  const float* g1  = (const float*)d_in[3];
  const float* b1  = (const float*)d_in[4];
  const float* w2  = (const float*)d_in[5];
  const float* g2  = (const float*)d_in[6];
  const float* b2  = (const float*)d_in[7];
  const float* wid = (const float*)d_in[8];
  const float* gid = (const float*)d_in[9];
  const float* bid = (const float*)d_in[10];
  float* out = (float*)d_out;

  const int CIN = 32, COUT = 64;
  const int N = in_sizes[0] / CIN;
  const int K = in_sizes[1] / N;   // 27

  // Workspace carving (harness poisons once; everything we read is written first)
  char* p = (char*)d_ws;
  auto carve = [&](size_t bytes) -> char* {
    char* r = p;
    p += (bytes + 255) & ~(size_t)255;
    return r;
  };
  _Float16* xh   = (_Float16*)carve((size_t)N * CIN * 2);   // x in f16
  float*    ybuf = (float*)   carve((size_t)N * COUT * 4);  // y1, later y2
  _Float16* hh   = (_Float16*)carve((size_t)N * COUT * 2);  // h in f16
  float*    idb  = (float*)   carve((size_t)N * COUT * 4);  // identity branch (pre-BN)
  _Float16* w1f  = (_Float16*)carve((size_t)K * 1 * 4 * 1024);
  _Float16* w2f  = (_Float16*)carve((size_t)K * 2 * 4 * 1024);
  _Float16* widf = (_Float16*)carve(4096);
  const int SBLK = 256;
  float* partA = (float*)carve((size_t)SBLK * 128 * 4);
  float* partB = (float*)carve((size_t)SBLK * 128 * 4);
  float* ab1  = (float*)carve(512);
  float* abid = (float*)carve(512);
  float* ab2  = (float*)carve(512);

  const float invN = 1.0f / (float)N;

  // 1) weight pre-swizzle into WMMA B-fragments
  {
    int t1 = K * 1 * 4 * 32;
    prep_frags<<<(t1 + 255) / 256, 256, 0, stream>>>(w1, w1f, K, 1);
    int t2 = K * 2 * 4 * 32;
    prep_frags<<<(t2 + 255) / 256, 256, 0, stream>>>(w2, w2f, K, 2);
    prep_frags<<<1, 128, 0, stream>>>(wid, widf, 1, 1);
  }
  // 2) cast x to f16 (halves the L2-resident gather footprint)
  {
    int tot4 = (N * CIN) / 4;
    f32_to_f16<<<(tot4 + 255) / 256, 256, 0, stream>>>(x, xh, tot4);
  }

  const int waves   = (N + 31) / 32;
  const int cblocks = (waves + 7) / 8;

  // 3) conv1 + identity GEMM (both WMMA)
  conv_wmma<1><<<cblocks, 256, 0, stream>>>(xh, CIN, nbr, K, w1f, ybuf, N);
  conv_wmma<1><<<cblocks, 256, 0, stream>>>(xh, CIN, nullptr, 1, widf, idb, N);

  // 4) BN stats (deterministic two-stage reduction)
  bn_stats_partial<<<SBLK, 256, 0, stream>>>(ybuf, partA, N);
  bn_stats_partial<<<SBLK, 256, 0, stream>>>(idb, partB, N);
  bn_finalize<<<1, 64, 0, stream>>>(partA, SBLK, g1, b1, ab1, invN);
  bn_finalize<<<1, 64, 0, stream>>>(partB, SBLK, gid, bid, abid, invN);

  // 5) h = f16(relu(bn1(y1)))
  const int tot4o = (N * COUT) / 4;
  bn_relu_f16<<<(tot4o + 255) / 256, 256, 0, stream>>>(ybuf, ab1, hh, tot4o);

  // 6) conv2 (WMMA), overwrite ybuf
  conv_wmma<2><<<cblocks, 256, 0, stream>>>(hh, COUT, nbr, K, w2f, ybuf, N);

  // 7) BN2 stats + final fused output
  bn_stats_partial<<<SBLK, 256, 0, stream>>>(ybuf, partA, N);
  bn_finalize<<<1, 64, 0, stream>>>(partA, SBLK, g2, b2, ab2, invN);
  final_out_k<<<(tot4o + 255) / 256, 256, 0, stream>>>(ybuf, idb, ab2, abid, out, tot4o);
}
